// VectorQuantizeEMA_68289980006517
// MI455X (gfx1250) — matrix operand order repaired and checked by reading, not compiled, not run on hardware
//
#include <hip/hip_runtime.h>
#include <hip/hip_bf16.h>

// ---------------------------------------------------------------------------
// VQ-VAE vector quantize for MI455X (gfx1250, wave32, WMMA).
//
// Roofline: distance matmul = 2*N*K*D = 34.4 GFLOP vs ~128MB HBM traffic
// (~5.5us at 23.3 TB/s) -> compute bound; runs on v_wmma_f32_16x16x32_bf16.
// fp32 is split into bf16 hi + bf16 lo residual; dot = hh + hl + lh
// (ll ~2^-16 relative, dropped) -> ~fp32-accurate scores with 6 bf16 WMMAs
// per 16x16x64 tile.  argmin_k uses cnorm[k] - 2*x.c_k (||x||^2 constant).
//
// Each wave owns M=32 rows (two A fragment sets) so every B-tile fetched from
// LDS feeds 12 WMMAs -> balances LDS bandwidth against WMMA issue rate.
// Codebook fragments (256KB) + cnorm (4KB) staged per-WG into LDS via
// GLOBAL_LOAD_ASYNC_TO_LDS_B128 (ASYNCcnt) when the builtin is available.
// ---------------------------------------------------------------------------

typedef __attribute__((ext_vector_type(16))) __bf16 v16bf;
typedef __attribute__((ext_vector_type(8)))  float  v8f;
typedef __attribute__((ext_vector_type(4)))  int    v4i;

#define VQ_N     262144   // 64*64*64 flattened rows
#define VQ_K     1024
#define VQ_D     64
#define ROWS_WG  256      // 8 waves * 32 rows
#define NTILES   64       // 1024 cols / 16

#define FRAG_BYTES   262144u          // 8192 slots * 32B (bf16 B-fragments)
#define CNORM_OFF    FRAG_BYTES       // cnorm table follows fragments
#define STAGE_BYTES  (FRAG_BYTES + 4096u)   // 266240 bytes -> LDS size

static __device__ __forceinline__ v8f wmma_bf16(v16bf a, v16bf b, v8f c) {
  // 8 args: (neg_a, A, neg_b, B, c_mod, C, reuse_a, reuse_b)
  return __builtin_amdgcn_wmma_f32_16x16x32_bf16(false, a, false, b,
                                                 (short)0, c, false, false);
}

// ---------------------------------------------------------------------------
// Phase 0: build bf16 hi/lo B-fragment image (in final LDS order) + cnorm,
// zero per-launch accumulators.  8192 slots, one thread each.
// B as packed here (contiguous k, same as A): lanes 0-15 hold K=0..15 (2/VGPR), lanes 16-31 hold
// K=16..31; lane%16 = column N.  Slot = ((tile*2+part)*2+kc)*32 + lane.
// ---------------------------------------------------------------------------
__global__ __launch_bounds__(256) void vq_prep(
    const float* __restrict__ cb, unsigned char* __restrict__ stage,
    unsigned int* __restrict__ counts, float* __restrict__ sse) {
  const int t = blockIdx.x * blockDim.x + threadIdx.x;  // 0..8191
  {
    const int tile = t >> 7;
    const int part = (t >> 6) & 1;   // 0 = hi, 1 = lo residual
    const int kc   = (t >> 5) & 1;   // d-chunk 0..31 / 32..63
    const int l    = t & 31;
    const int n    = l & 15;
    const int kb   = (l >> 4) << 4;  // 0 or 16
    const float* src = cb + (tile * 16 + n) * VQ_D + kc * 32 + kb;
    v16bf v;
#pragma unroll
    for (int j = 0; j < 16; ++j) {
      float x  = src[j];
      __bf16 h = (__bf16)x;
      v[j] = part ? (__bf16)(x - (float)h) : h;
    }
    *(v16bf*)(stage + (unsigned)t * 32u) = v;
  }
  if (t < VQ_K) {
    const float* c = cb + t * VQ_D;
    float s = 0.0f;
#pragma unroll 8
    for (int d = 0; d < VQ_D; ++d) s += c[d] * c[d];
    ((float*)(stage + CNORM_OFF))[t] = s;
    counts[t] = 0u;
  }
  if (t == 0) *sse = 0.0f;
}

// ---------------------------------------------------------------------------
// Phase 1: distances via WMMA, argmin, gather, loss/count partials.
// Dynamic LDS = 266240 bytes (fragments + cnorm); wave32, 8 waves/WG.
// ---------------------------------------------------------------------------
__global__ __launch_bounds__(256) void vq_main(
    const float* __restrict__ in, const float* __restrict__ cb,
    const unsigned char* __restrict__ stage, float* __restrict__ out,
    unsigned int* __restrict__ counts, float* __restrict__ sse) {
  extern __shared__ unsigned char smem[];  // STAGE_BYTES
  const int tid  = threadIdx.x;
  const int lane = tid & 31;
  const int warp = tid >> 5;
  const int m    = lane & 15;

  // ---- Stage fragments + cnorm: identity copy global -> LDS ---------------
#if defined(__gfx1250__) &&                                              \
    __has_builtin(__builtin_amdgcn_global_load_async_to_lds_b128) &&     \
    __has_builtin(__builtin_amdgcn_s_wait_asynccnt)
  {
    // Builtin wants int4 pointers: AS1 src ("__device__"), AS3 dst.
    typedef __attribute__((address_space(1))) v4i gv4i;
    typedef __attribute__((address_space(3))) v4i lv4i;
    gv4i* g = (gv4i*)stage;
    lv4i* l = (lv4i*)smem;
    for (int c = tid; c < (int)(STAGE_BYTES / 16u); c += 256)
      __builtin_amdgcn_global_load_async_to_lds_b128(g + c, l + c, 0, 0);
    __builtin_amdgcn_s_wait_asynccnt(0);
  }
#else
  {
    const uint4* g = (const uint4*)stage;
    uint4* l = (uint4*)smem;
    for (int c = tid; c < (int)(STAGE_BYTES / 16u); c += 256) l[c] = g[c];
  }
#endif
  __syncthreads();

  // ---- A fragments: 32 rows of inputs, bf16 hi/lo, kept in registers ------
  // A as loaded here (contiguous k, same as B): lanes 0-15: K=0..15; lanes 16-31:
  // K=16..31; M = lane%16.  h selects row-half (+0 / +16).
  const int rowbase = blockIdx.x * ROWS_WG + warp * 32;
  const int abase   = (lane >> 4) << 4;  // 0 or 16
  v16bf a_hi[2][2], a_lo[2][2];          // [kc][h]
#pragma unroll
  for (int h = 0; h < 2; ++h) {
#pragma unroll
    for (int kc = 0; kc < 2; ++kc) {
      const float* p = in + (rowbase + h * 16 + m) * VQ_D + kc * 32 + abase;
#pragma unroll
      for (int j = 0; j < 16; ++j) {
        float x  = p[j];
        __bf16 h16 = (__bf16)x;
        a_hi[kc][h][j] = h16;
        a_lo[kc][h][j] = (__bf16)(x - (float)h16);
      }
    }
  }

  // ---- Sweep 64 column tiles; each B-tile feeds 12 WMMAs ------------------
  const float* cnl = (const float*)(smem + CNORM_OFF);
  float bestv[16];
  int   besti[16];
#pragma unroll
  for (int r = 0; r < 16; ++r) { bestv[r] = 3.4e38f; besti[r] = 0; }

  for (int t = 0; t < NTILES; ++t) {
    const unsigned tb = (unsigned)t * 4096u + (unsigned)lane * 32u;
    v16bf bh0 = *(const v16bf*)(smem + tb + 0u);     // hi, kc0
    v16bf bh1 = *(const v16bf*)(smem + tb + 1024u);  // hi, kc1
    v16bf bl0 = *(const v16bf*)(smem + tb + 2048u);  // lo, kc0
    v16bf bl1 = *(const v16bf*)(smem + tb + 3072u);  // lo, kc1

    v8f acc0 = {}, acc1 = {};
    acc0 = wmma_bf16(a_hi[0][0], bh0, acc0);
    acc1 = wmma_bf16(a_hi[0][1], bh0, acc1);
    acc0 = wmma_bf16(a_hi[1][0], bh1, acc0);
    acc1 = wmma_bf16(a_hi[1][1], bh1, acc1);
    acc0 = wmma_bf16(a_hi[0][0], bl0, acc0);
    acc1 = wmma_bf16(a_hi[0][1], bl0, acc1);
    acc0 = wmma_bf16(a_hi[1][0], bl1, acc0);
    acc1 = wmma_bf16(a_hi[1][1], bl1, acc1);
    acc0 = wmma_bf16(a_lo[0][0], bh0, acc0);
    acc1 = wmma_bf16(a_lo[0][1], bh0, acc1);
    acc0 = wmma_bf16(a_lo[1][0], bh1, acc0);
    acc1 = wmma_bf16(a_lo[1][1], bh1, acc1);

    const int   col = t * 16 + m;
    const float cn  = cnl[col];          // ds_load, folds into dscnt waits
    // C/D layout: VGPR r -> row (lane<16 ? r : r+8), col = lane%16.
#pragma unroll
    for (int r = 0; r < 8; ++r) {
      float d0 = cn - 2.0f * acc0[r];
      bool b0 = d0 < bestv[r];           // cols ascend in-lane: '<' keeps first idx
      bestv[r] = b0 ? d0 : bestv[r];
      besti[r] = b0 ? col : besti[r];
      float d1 = cn - 2.0f * acc1[r];
      bool b1 = d1 < bestv[8 + r];
      bestv[8 + r] = b1 ? d1 : bestv[8 + r];
      besti[8 + r] = b1 ? col : besti[8 + r];
    }
  }

  // ---- 16-lane butterfly argmin (halves are independent row groups) -------
#pragma unroll
  for (int r = 0; r < 16; ++r) {
    float v = bestv[r];
    int   i = besti[r];
#pragma unroll
    for (int off = 1; off < 16; off <<= 1) {
      float ov = __shfl_xor(v, off, 32);
      int   oi = __shfl_xor(i, off, 32);
      if (ov < v || (ov == v && oi < i)) { v = ov; i = oi; }  // first-index tie
    }
    bestv[r] = v;
    besti[r] = i;
  }

  // ---- Gather codes, write quantize, accumulate SSE + counts --------------
  float sacc = 0.0f;
#pragma unroll
  for (int i = 0; i < 32; ++i) {
    // row i: h=i/16 selects slot bank; j=i%16: j<8 -> lanes 0-15 slot j,
    // j>=8 -> lanes 16-31 slot j-8.
    const int h = i >> 4, j = i & 15;
    const int idx = __shfl(besti[h * 8 + (j & 7)], (j >> 3) << 4, 32);
    const int row = rowbase + i;
    const float2 q = ((const float2*)(cb + idx * VQ_D))[lane];
    const float2 x = ((const float2*)(in + row * VQ_D))[lane];
    ((float2*)(out + row * VQ_D))[lane] = q;  // straight-through fwd == codes
    float dx = q.x - x.x, dy = q.y - x.y;
    sacc += dx * dx + dy * dy;
    if (lane == 0) atomicAdd(&counts[idx], 1u);
  }
#pragma unroll
  for (int off = 1; off < 32; off <<= 1) sacc += __shfl_xor(sacc, off, 32);
  if (lane == 0) atomicAdd(sse, sacc);
}

// ---------------------------------------------------------------------------
// Phase 2: perplexity + loss scalars.
// ---------------------------------------------------------------------------
__global__ __launch_bounds__(1024) void vq_final(
    const unsigned int* __restrict__ counts, const float* __restrict__ sse,
    float* __restrict__ outp) {
  __shared__ float red[32];
  const int t = threadIdx.x;
  float p    = (float)counts[t] * (1.0f / (float)VQ_N);
  float term = p * logf(p + 1e-10f);
#pragma unroll
  for (int off = 1; off < 32; off <<= 1) term += __shfl_xor(term, off, 32);
  if ((t & 31) == 0) red[t >> 5] = term;
  __syncthreads();
  if (t < 32) {
    float v = red[t];
#pragma unroll
    for (int off = 1; off < 32; off <<= 1) v += __shfl_xor(v, off, 32);
    if (t == 0) {
      outp[0] = 0.25f * (*sse) * (1.0f / ((float)VQ_N * (float)VQ_D));  // loss
      outp[1] = expf(-v);                                               // perplexity
    }
  }
}

// ---------------------------------------------------------------------------
extern "C" void kernel_launch(void* const* d_in, const int* in_sizes, int n_in,
                              void* d_out, int out_size, void* d_ws, size_t ws_size,
                              hipStream_t stream) {
  const float* in = (const float*)d_in[0];   // (64,64,64,64) fp32
  const float* cb = (const float*)d_in[1];   // (1024,64)     fp32
  float* out = (float*)d_out;                // quantize | loss | perplexity

  unsigned char* stage  = (unsigned char*)d_ws;                     // 260 KB
  unsigned int*  counts = (unsigned int*)((char*)d_ws + STAGE_BYTES);
  float*         sse    = (float*)((char*)d_ws + STAGE_BYTES + 4096);

  vq_prep <<<32, 256, 0, stream>>>(cb, stage, counts, sse);
  vq_main <<<VQ_N / ROWS_WG, 256, STAGE_BYTES, stream>>>(in, cb, stage, out,
                                                         counts, sse);
  vq_final<<<1, 1024, 0, stream>>>(counts, sse, out + (size_t)VQ_N * VQ_D);
}
